// CoarseMatching_70102456206132
// MI455X (gfx1250) — matrix operand order repaired and checked by reading, not compiled, not run on hardware
//
#include <hip/hip_runtime.h>

typedef __attribute__((ext_vector_type(16))) _Float16 v16h;
typedef __attribute__((ext_vector_type(8)))  _Float16 v8h;
typedef __attribute__((ext_vector_type(8)))  float    v8f;

#define NB    2
#define LDIM  4800
#define SDIM  4800
#define CDIM  256
#define MARGIN 2
#define SVEC  (SDIM / 4)   // 1200 float4 per row

__device__ __constant__ float kInvScale = 1.0f / (256.0f * 0.1f); // 1/(C*TEMP)

// ---------------------------------------------------------------- f32 -> f16
__global__ void cvt_f16_kernel(const float* __restrict__ src,
                               _Float16* __restrict__ dst, int n) {
  int i = blockIdx.x * blockDim.x + threadIdx.x;
  if (i < n) dst[i] = (_Float16)src[i];
}

// ---------------------------------------------------------------- WMMA GEMM
// sim[n][l][s] = sum_c f0[n][l][c] * f1[n][s][c] * kInvScale
// block = 128 threads (4 waves), block tile 96x96, wave tile 48x48 (3x3 WMMA)
__global__ __launch_bounds__(128)
void sim_gemm_kernel(const _Float16* __restrict__ A,   // [NB, LDIM, CDIM] f16
                     const _Float16* __restrict__ B,   // [NB, SDIM, CDIM] f16
                     float* __restrict__ sim) {        // [NB, LDIM, SDIM]
  const int lane = threadIdx.x & 31;
  const int wave = threadIdx.x >> 5;          // 0..3
  const int wy   = wave >> 1, wx = wave & 1;  // 2x2 wave grid
  const int n    = blockIdx.z;
  const int mbase = blockIdx.x * 96 + wy * 48;
  const int sbase = blockIdx.y * 96 + wx * 48;
  const int half  = lane >> 4;                // lane half (0/1)
  const int lr    = lane & 15;

  const _Float16* __restrict__ Ab = A + (size_t)n * LDIM * CDIM;
  const _Float16* __restrict__ Bb = B + (size_t)n * SDIM * CDIM;

  v8f acc[3][3] = {};

#pragma unroll 1
  for (int k0 = 0; k0 < CDIM; k0 += 32) {
    v16h af[3], bf[3];
#pragma unroll
    for (int t = 0; t < 3; ++t) {
      // A frag (16-bit A 16x32): lane half 0 -> K 0..7 & 16..23, half 1 -> +8
      const _Float16* ap = Ab + (size_t)(mbase + t * 16 + lr) * CDIM + k0 + half * 8;
      v8h alo = *(const v8h*)(ap);
      v8h ahi = *(const v8h*)(ap + 16);
#pragma unroll
      for (int e = 0; e < 8; ++e) { af[t][e] = alo[e]; af[t][8 + e] = ahi[e]; }
      // B frag (16-bit B 32x16): lane half 0 -> K 0..7 & 16..23, half 1 -> +8
      const _Float16* bp = Bb + (size_t)(sbase + t * 16 + lr) * CDIM + k0 + half * 8;
      v8h blo = *(const v8h*)(bp);
      v8h bhi = *(const v8h*)(bp + 16);
#pragma unroll
      for (int e = 0; e < 8; ++e) { bf[t][e] = blo[e]; bf[t][8 + e] = bhi[e]; }
    }
#pragma unroll
    for (int i = 0; i < 3; ++i)
#pragma unroll
      for (int j = 0; j < 3; ++j)
        acc[i][j] = __builtin_amdgcn_wmma_f32_16x16x32_f16(
            false, af[i], false, bf[j], (short)0, acc[i][j], false, false);
  }

  const float scale = kInvScale;
#pragma unroll
  for (int i = 0; i < 3; ++i)
#pragma unroll
    for (int j = 0; j < 3; ++j) {
      const int col = sbase + j * 16 + lr;
#pragma unroll
      for (int v = 0; v < 8; ++v) {
        const int row = mbase + i * 16 + v + 8 * half;  // C/D layout: VGPR v, lane half
        sim[((size_t)n * LDIM + row) * SDIM + col] = acc[i][j][v] * scale;
      }
    }
}

// ---------------------------------------------------------------- reductions
// conf expression MUST be identical in every kernel so bitwise == holds for
// the mutual-nearest-neighbor test (fmaxf returns one of its inputs bitwise).
__device__ __forceinline__ float conf_of(float simv, float rs, float cs) {
  float e = __expf(simv);               // v_exp_f32 path (memory-bound passes)
  return __fdividef(e * e, rs * cs);    // fast rcp-based divide, deterministic
}

__global__ void row_sum_kernel(const float* __restrict__ sim,
                               float* __restrict__ rowsum) {
  __shared__ float red[256];
  const int row = blockIdx.x;                 // n*LDIM + l
  const float4* p = (const float4*)(sim + (size_t)row * SDIM);
  float s = 0.f;
  for (int i = threadIdx.x; i < SVEC; i += 256) {
    float4 v = p[i];
    s += __expf(v.x) + __expf(v.y) + __expf(v.z) + __expf(v.w);
  }
  red[threadIdx.x] = s; __syncthreads();
  for (int off = 128; off > 0; off >>= 1) {
    if (threadIdx.x < off) red[threadIdx.x] += red[threadIdx.x + off];
    __syncthreads();
  }
  if (threadIdx.x == 0) rowsum[row] = red[0];
}

__global__ void col_sum_kernel(const float* __restrict__ sim,
                               float* __restrict__ colsum) {
  const int n = blockIdx.y;
  const int s = blockIdx.x * 256 + threadIdx.x;
  if (s >= SDIM) return;
  const float* p = sim + (size_t)n * LDIM * SDIM + s;
  float acc = 0.f;
  for (int l = 0; l < LDIM; ++l) acc += __expf(p[(size_t)l * SDIM]);
  colsum[n * SDIM + s] = acc;
}

__global__ void row_max_kernel(const float* __restrict__ sim,
                               const float* __restrict__ rowsum,
                               const float* __restrict__ colsum,
                               float* __restrict__ rowmax) {
  __shared__ float red[256];
  const int row = blockIdx.x;
  const int n   = row / LDIM;
  const float rs = rowsum[row];
  const float4* p  = (const float4*)(sim + (size_t)row * SDIM);
  const float4* cs = (const float4*)(colsum + n * SDIM);
  float m = 0.f;
  for (int i = threadIdx.x; i < SVEC; i += 256) {
    float4 v = p[i], c = cs[i];
    m = fmaxf(m, conf_of(v.x, rs, c.x));
    m = fmaxf(m, conf_of(v.y, rs, c.y));
    m = fmaxf(m, conf_of(v.z, rs, c.z));
    m = fmaxf(m, conf_of(v.w, rs, c.w));
  }
  red[threadIdx.x] = m; __syncthreads();
  for (int off = 128; off > 0; off >>= 1) {
    if (threadIdx.x < off) red[threadIdx.x] = fmaxf(red[threadIdx.x], red[threadIdx.x + off]);
    __syncthreads();
  }
  if (threadIdx.x == 0) rowmax[row] = red[0];
}

__global__ void col_max_kernel(const float* __restrict__ sim,
                               const float* __restrict__ rowsum,
                               const float* __restrict__ colsum,
                               float* __restrict__ colmax) {
  const int n = blockIdx.y;
  const int s = blockIdx.x * 256 + threadIdx.x;
  if (s >= SDIM) return;
  const float cs = colsum[n * SDIM + s];
  const float* p   = sim + (size_t)n * LDIM * SDIM + s;
  const float* rsb = rowsum + n * LDIM;
  float m = 0.f;
  for (int l = 0; l < LDIM; ++l) m = fmaxf(m, conf_of(p[(size_t)l * SDIM], rsb[l], cs));
  colmax[n * SDIM + s] = m;
}

// ------------------------------------------------- interior-margin flag tables
__global__ void interior_kernel(const int* __restrict__ ph, const int* __restrict__ pw,
                                unsigned char* __restrict__ flags, int len) {
  int i = blockIdx.x * 256 + threadIdx.x;
  if (i >= len) return;
  const int h = *ph, w = *pw;
  const int r = i / w, c = i % w;
  flags[i] = (r >= MARGIN && r < h - MARGIN && c >= MARGIN && c < w - MARGIN) ? 1 : 0;
}

// ---------------------------------------------------------------- finalize
// grid = (SDIM/256, LDIM, NB): free index decomposition, no 64-bit div/mod.
// out[0..NLS)     : sim (read) -> conf (written in place)
// out[NLS..2NLS)  : mask as 0/1 float
// out[2NLS..3NLS) : matched_conf
__global__ void finalize_kernel(float* __restrict__ out,
                                const float* __restrict__ rowsum,
                                const float* __restrict__ colsum,
                                const float* __restrict__ rowmax,
                                const float* __restrict__ colmax,
                                const unsigned char* __restrict__ int0,
                                const unsigned char* __restrict__ int1) {
  const int s = blockIdx.x * 256 + threadIdx.x;
  if (s >= SDIM) return;
  const int l = blockIdx.y;
  const int n = blockIdx.z;
  const size_t NLS = (size_t)NB * LDIM * SDIM;
  const size_t r   = (size_t)n * LDIM + l;
  const size_t idx = r * SDIM + s;

  const float simv = out[idx];
  const float c = conf_of(simv, rowsum[r], colsum[(size_t)n * SDIM + s]);
  const bool mask = (c > 0.2f) && (int0[l] & int1[s]) &&
                    (c == rowmax[r]) && (c == colmax[(size_t)n * SDIM + s]);
  out[idx]           = c;
  out[NLS + idx]     = mask ? 1.0f : 0.0f;
  out[2 * NLS + idx] = mask ? c : 0.0f;
}

// ---------------------------------------------------------------- launch
extern "C" void kernel_launch(void* const* d_in, const int* in_sizes, int n_in,
                              void* d_out, int out_size, void* d_ws, size_t ws_size,
                              hipStream_t stream) {
  (void)in_sizes; (void)n_in; (void)out_size; (void)ws_size;
  const float* f0 = (const float*)d_in[0];
  const float* f1 = (const float*)d_in[1];
  const int* ph0 = (const int*)d_in[2];
  const int* pw0 = (const int*)d_in[3];
  const int* ph1 = (const int*)d_in[4];
  const int* pw1 = (const int*)d_in[5];
  float* out = (float*)d_out;

  const int elemsF = NB * LDIM * CDIM;             // 2,457,600 per feature tensor
  _Float16* h0buf = (_Float16*)d_ws;
  _Float16* h1buf = h0buf + (size_t)elemsF;
  float* rowsum = (float*)(h1buf + (size_t)elemsF);
  float* colsum = rowsum + NB * LDIM;
  float* rowmax = colsum + NB * SDIM;
  float* colmax = rowmax + NB * LDIM;
  unsigned char* int0 = (unsigned char*)(colmax + NB * SDIM);
  unsigned char* int1 = int0 + LDIM;

  float* sim = out;   // sim lives in the conf slot of d_out, overwritten in finalize

  cvt_f16_kernel<<<(elemsF + 255) / 256, 256, 0, stream>>>(f0, h0buf, elemsF);
  cvt_f16_kernel<<<(elemsF + 255) / 256, 256, 0, stream>>>(f1, h1buf, elemsF);

  sim_gemm_kernel<<<dim3(LDIM / 96, SDIM / 96, NB), 128, 0, stream>>>(h0buf, h1buf, sim);

  row_sum_kernel<<<NB * LDIM, 256, 0, stream>>>(sim, rowsum);
  col_sum_kernel<<<dim3((SDIM + 255) / 256, NB), 256, 0, stream>>>(sim, colsum);
  row_max_kernel<<<NB * LDIM, 256, 0, stream>>>(sim, rowsum, colsum, rowmax);
  col_max_kernel<<<dim3((SDIM + 255) / 256, NB), 256, 0, stream>>>(sim, rowsum, colsum, colmax);

  interior_kernel<<<(LDIM + 255) / 256, 256, 0, stream>>>(ph0, pw0, int0, LDIM);
  interior_kernel<<<(SDIM + 255) / 256, 256, 0, stream>>>(ph1, pw1, int1, SDIM);

  finalize_kernel<<<dim3((SDIM + 255) / 256, LDIM, NB), 256, 0, stream>>>(
      out, rowsum, colsum, rowmax, colmax, int0, int1);
}
